// NetGram_21345987461490
// MI455X (gfx1250) — matrix operand order repaired and checked by reading, not compiled, not run on hardware
//
#include <hip/hip_runtime.h>

// ---------------------------------------------------------------------------
// Sizes
// ---------------------------------------------------------------------------
#define BATCH     8192
#define IN_DIM    784
#define KPAD      800          // 25 * 32
#define KITERS    25
#define NNETS     30
#define NNODES    20
#define NCOLS     600          // NNETS * NNODES
#define NTILE_PAD 40           // col tiles padded so NT divides evenly (640 cols)
#define NUM_OUT   10
#define DIM_OUT   16

// register blocking: each wave owns MT x NT 16x16 tiles
#define MT        2
#define NT        4
#define BMGROUPS  (512 / MT)       // 256 row-tile groups
#define JGROUPS   (NTILE_PAD / NT) // 10  col-tile groups

typedef __attribute__((ext_vector_type(16))) __bf16 v16bf;
typedef __attribute__((ext_vector_type(8)))  __bf16 v8bf;
typedef __attribute__((ext_vector_type(8)))  float  v8f;

__device__ __forceinline__ unsigned short f32_to_bf16(float f) {
    unsigned int u = __float_as_uint(f);
    u += 0x7FFFu + ((u >> 16) & 1u);     // round-to-nearest-even
    return (unsigned short)(u >> 16);
}

// ---------------------------------------------------------------------------
// Pack x [8192,784] f32 -> bf16 [8192,800] (zero-padded K)
// ---------------------------------------------------------------------------
__global__ __launch_bounds__(256) void pack_x(const float* __restrict__ x,
                                              unsigned short* __restrict__ xb) {
    int idx = blockIdx.x * 256 + threadIdx.x;          // < 8192*800
    int b = idx / KPAD;
    int k = idx - b * KPAD;
    unsigned short v = 0;
    if (k < IN_DIM) v = f32_to_bf16(x[(size_t)b * IN_DIM + k]);
    xb[idx] = v;
}

// ---------------------------------------------------------------------------
// Pack W1 [30,784,20] f32 into per-lane B-fragment layout:
//   wp[((jn*25 + kIter)*32 + lane)*16 + t] = B[k][j] as bf16
//   k = kIter*32 + (lane<16 ? 0 : 8) + (t<8 ? t : 8 + t),  j = jn*16 + (lane&15)
//   B[k][j] = W1[n=j/20][k][d=j%20], zero outside bounds.
// ---------------------------------------------------------------------------
__global__ __launch_bounds__(256) void pack_w1(const float* __restrict__ W1,
                                               unsigned short* __restrict__ wp) {
    int idx = blockIdx.x * 256 + threadIdx.x;          // < 40*25*32*16 = 512000
    int t     = idx & 15;
    int lane  = (idx >> 4) & 31;
    int kIter = (idx >> 9) % KITERS;
    int jn    = idx / (16 * 32 * KITERS);
    int k = kIter * 32 + ((lane < 16) ? 0 : 8) + ((t < 8) ? t : 8 + t);
    int j = jn * 16 + (lane & 15);
    unsigned short v = 0;
    if (k < IN_DIM && j < NCOLS) {
        int n = j / NNODES, d = j - n * NNODES;
        v = f32_to_bf16(W1[((size_t)n * IN_DIM + k) * NNODES + d]);
    }
    wp[idx] = v;
}

// ---------------------------------------------------------------------------
// GEMM1: h[b,j] = relu( sum_k xb[b,k]*B[k,j] + b1[j] )
// Each wave: MTxNT block of 16x16 tiles -> 8 WMMAs per k-step on 12 b128 loads.
// ---------------------------------------------------------------------------
__global__ __launch_bounds__(128) void gemm1_wmma(const unsigned short* __restrict__ xb,
                                                  const unsigned short* __restrict__ wp,
                                                  const float* __restrict__ b1,
                                                  float* __restrict__ h) {
    const int lane = threadIdx.x & 31;
    const int wave = threadIdx.x >> 5;
    const int wid  = blockIdx.x * 4 + wave;           // < BMGROUPS*JGROUPS = 2560
    const int bmg  = wid / JGROUPS;
    const int jng  = wid - bmg * JGROUPS;
    const int bm0  = bmg * MT;                        // first row tile
    const int jn0  = jng * NT;                        // first col tile

    // A fragments: row = (bm0+m)*16 + (lane&15); chunks at k0+aoff and +16
    const unsigned short* xrow = xb + (size_t)(bm0 * 16 + (lane & 15)) * KPAD
                                    + ((lane < 16) ? 0 : 8);
    // B fragments: contiguous 16 bf16 per lane; tile stride 25*32*16 elements
    const unsigned short* wl = wp + ((size_t)jn0 * KITERS * 32 + lane) * 16;

    v8f c[MT][NT];
    #pragma unroll
    for (int m = 0; m < MT; ++m)
        #pragma unroll
        for (int j = 0; j < NT; ++j) c[m][j] = (v8f){};

    #pragma unroll 5
    for (int it = 0; it < KITERS; ++it) {
        const int k0 = it * 32;
        v16bf a[MT];
        #pragma unroll
        for (int m = 0; m < MT; ++m) {
            const unsigned short* p = xrow + (size_t)m * 16 * KPAD + k0;
            v8bf alo = *(const v8bf*)(const void*)(p);
            v8bf ahi = *(const v8bf*)(const void*)(p + 16);
            #pragma unroll
            for (int i = 0; i < 8; ++i) { a[m][i] = alo[i]; a[m][8 + i] = ahi[i]; }
        }
        v16bf bfrag[NT];
        #pragma unroll
        for (int j = 0; j < NT; ++j)
            bfrag[j] = *(const v16bf*)(const void*)(wl + (size_t)j * KITERS * 32 * 16
                                                       + (size_t)it * 32 * 16);
        #pragma unroll
        for (int m = 0; m < MT; ++m)
            #pragma unroll
            for (int j = 0; j < NT; ++j)
                c[m][j] = __builtin_amdgcn_wmma_f32_16x16x32_bf16(
                    false, a[m], false, bfrag[j], (short)0, c[m][j], false, false);
    }

    // C layout: VGPR v -> M = v + (lane>=16 ? 8 : 0), N = lane%16
    const int rb = (lane >= 16) ? 8 : 0;
    #pragma unroll
    for (int j = 0; j < NT; ++j) {
        const int col = (jn0 + j) * 16 + (lane & 15);
        if (col < NCOLS) {
            const float bias = b1[col];
            #pragma unroll
            for (int m = 0; m < MT; ++m) {
                const int rowbase = (bm0 + m) * 16 + rb;
                #pragma unroll
                for (int v = 0; v < 8; ++v) {
                    float val = c[m][j][v] + bias;
                    h[(size_t)(rowbase + v) * NCOLS + col] = val > 0.f ? val : 0.f;
                }
            }
        }
    }
}

// ---------------------------------------------------------------------------
// Fused tail: layer2 + squash + priors + 3-iter dynamic routing.
// 1 wave per batch element, 4 per block. All intermediates in LDS.
// ---------------------------------------------------------------------------
#define BPB 4
__global__ __launch_bounds__(128) void capsule_tail(const float* __restrict__ h,
                                                    const float* __restrict__ W2,
                                                    const float* __restrict__ b2,
                                                    const float* __restrict__ rw,
                                                    float* __restrict__ out) {
    __shared__ float s_u  [BPB][NCOLS];                    // 600
    __shared__ float s_pri[BPB][NUM_OUT * NNETS * DIM_OUT];// 4800
    __shared__ float s_log[BPB][NUM_OUT * NNETS];          // 300
    __shared__ float s_prb[BPB][NUM_OUT * NNETS];          // 300
    __shared__ float s_sv [BPB][NUM_OUT * DIM_OUT];        // 160
    __shared__ float s_scl[BPB][32];

    const int lane = threadIdx.x & 31;
    const int w    = threadIdx.x >> 5;
    const int b    = blockIdx.x * BPB + w;
    const float* hrow = h + (size_t)b * NCOLS;

    // ---- layer 2: h2[n,e] = relu(sum_d h[n,d]*W2[n,d,e] + b2[n,e]) ----
    for (int idx = lane; idx < NCOLS; idx += 32) {
        int n = idx / NNODES, e = idx - n * NNODES;
        float acc = b2[idx];
        const float* hn = hrow + n * NNODES;
        const float* w2 = W2 + (size_t)n * NNODES * NNODES + e;
        #pragma unroll
        for (int d = 0; d < NNODES; ++d) acc += hn[d] * w2[(size_t)d * NNODES];
        s_u[w][idx] = acc > 0.f ? acc : 0.f;
    }
    __syncthreads();

    // ---- squash over features per net ----
    for (int n = lane; n < NNETS; n += 32) {
        float sq = 0.f;
        #pragma unroll
        for (int d = 0; d < NNODES; ++d) { float t = s_u[w][n * NNODES + d]; sq += t * t; }
        s_scl[w][n] = sq / ((1.f + sq) * sqrtf(sq));
    }
    __syncthreads();
    for (int idx = lane; idx < NCOLS; idx += 32) s_u[w][idx] *= s_scl[w][idx / NNODES];
    __syncthreads();

    // ---- priors[o,n,k] = sum_d u[n,d] * rw[o,n,d,k] ----
    for (int idx = lane; idx < NUM_OUT * NNETS * DIM_OUT; idx += 32) {
        int o = idx / (NNETS * DIM_OUT);
        int r = idx - o * (NNETS * DIM_OUT);
        int n = r / DIM_OUT, k = r - n * DIM_OUT;
        const float* rwp = rw + (((size_t)o * NNETS + n) * NNODES) * DIM_OUT + k;
        const float* un  = &s_u[w][n * NNODES];
        float acc = 0.f;
        #pragma unroll
        for (int d = 0; d < NNODES; ++d) acc += un[d] * rwp[(size_t)d * DIM_OUT];
        s_pri[w][idx] = acc;
    }
    for (int idx = lane; idx < NUM_OUT * NNETS; idx += 32) s_log[w][idx] = 0.f;
    __syncthreads();

    // ---- routing iterations ----
    for (int iter = 0; iter < 3; ++iter) {
        // softmax over output capsules (o) for each net n
        for (int n = lane; n < NNETS; n += 32) {
            float m = -1e30f;
            #pragma unroll
            for (int o = 0; o < NUM_OUT; ++o) { float l = s_log[w][o * NNETS + n]; m = l > m ? l : m; }
            float sum = 0.f;
            #pragma unroll
            for (int o = 0; o < NUM_OUT; ++o) sum += __expf(s_log[w][o * NNETS + n] - m);
            float inv = 1.f / sum;
            #pragma unroll
            for (int o = 0; o < NUM_OUT; ++o)
                s_prb[w][o * NNETS + n] = __expf(s_log[w][o * NNETS + n] - m) * inv;
        }
        __syncthreads();
        // s[o,k] = sum_n probs[o,n] * priors[o,n,k]
        for (int idx = lane; idx < NUM_OUT * DIM_OUT; idx += 32) {
            int o = idx / DIM_OUT, k = idx - o * DIM_OUT;
            float acc = 0.f;
            #pragma unroll
            for (int n = 0; n < NNETS; ++n)
                acc += s_prb[w][o * NNETS + n] * s_pri[w][(o * NNETS + n) * DIM_OUT + k];
            s_sv[w][idx] = acc;
        }
        __syncthreads();
        // squash over k per o
        for (int o = lane; o < NUM_OUT; o += 32) {
            float sq = 0.f;
            #pragma unroll
            for (int k = 0; k < DIM_OUT; ++k) { float t = s_sv[w][o * DIM_OUT + k]; sq += t * t; }
            s_scl[w][o] = sq / ((1.f + sq) * sqrtf(sq));
        }
        __syncthreads();
        for (int idx = lane; idx < NUM_OUT * DIM_OUT; idx += 32)
            s_sv[w][idx] *= s_scl[w][idx / DIM_OUT];
        __syncthreads();
        // logits += sum_k priors[o,n,k] * v[o,k]   (not on last iteration)
        if (iter < 2) {
            for (int idx = lane; idx < NUM_OUT * NNETS; idx += 32) {
                int o = idx / NNETS, n = idx - o * NNETS;
                float acc = 0.f;
                #pragma unroll
                for (int k = 0; k < DIM_OUT; ++k)
                    acc += s_pri[w][(o * NNETS + n) * DIM_OUT + k] * s_sv[w][o * DIM_OUT + k];
                s_log[w][idx] += acc;
            }
        }
        __syncthreads();
    }

    // ---- output v[b,o,k] ----
    for (int idx = lane; idx < NUM_OUT * DIM_OUT; idx += 32)
        out[(size_t)b * NUM_OUT * DIM_OUT + idx] = s_sv[w][idx];
}

// ---------------------------------------------------------------------------
// Launch
// ---------------------------------------------------------------------------
extern "C" void kernel_launch(void* const* d_in, const int* in_sizes, int n_in,
                              void* d_out, int out_size, void* d_ws, size_t ws_size,
                              hipStream_t stream) {
    const float* x  = (const float*)d_in[0];
    const float* W1 = (const float*)d_in[1];
    const float* b1 = (const float*)d_in[2];
    const float* W2 = (const float*)d_in[3];
    const float* b2 = (const float*)d_in[4];
    const float* rw = (const float*)d_in[5];
    float* out = (float*)d_out;

    // workspace carve (bytes): xb 13,107,200 | wp 1,024,000 | h 19,660,800
    unsigned short* xb = (unsigned short*)d_ws;
    unsigned short* wp = xb + (size_t)BATCH * KPAD;
    float*          h  = (float*)(wp + (size_t)NTILE_PAD * KITERS * 32 * 16);

    pack_x <<<(BATCH * KPAD) / 256, 256, 0, stream>>>(x, xb);
    pack_w1<<<(NTILE_PAD * KITERS * 32 * 16) / 256, 256, 0, stream>>>(W1, wp);
    gemm1_wmma<<<(BMGROUPS * JGROUPS) / 4, 128, 0, stream>>>(xb, wp, b1, h);
    capsule_tail<<<BATCH / BPB, 128, 0, stream>>>(h, W2, b2, rw, out);
}
